// InfomaxEncoder_28587302322456
// MI455X (gfx1250) — matrix operand order repaired and checked by reading, not compiled, not run on hardware
//
#include <hip/hip_runtime.h>
#include <math.h>

// ---------------- types for WMMA fragments ----------------
typedef __attribute__((ext_vector_type(16))) _Float16 v16h;
typedef __attribute__((ext_vector_type(8)))  float    v8f;
typedef __attribute__((ext_vector_type(4)))  unsigned int u32x4;

union AF { v16h v; u32x4 u[2]; };

#define HD 128      // hidden dim
#define RB 50       // num rbf
#define KP 192      // padded K for GEMM1 (128 + 64)
#define NL 4        // layers
#define NM 64       // molecules
#define G3 384      // 3*H
#define CUT 5.0f
#define LN_EPS 1e-5f

// ---------------- small utility kernels ----------------
__global__ void zero_f32(float* __restrict__ p, int n) {
    int i = blockIdx.x * blockDim.x + threadIdx.x;
    if (i < n) p[i] = 0.0f;
}

__global__ void embed_kernel(const int* __restrict__ an, const float* __restrict__ emb,
                             float* __restrict__ x, int N) {
    int i = blockIdx.x * blockDim.x + threadIdx.x;
    if (i >= N * HD) return;
    int n = i / HD, h = i % HD;
    int a = an[n]; a = a < 0 ? 0 : (a > 99 ? 99 : a);
    x[i] = emb[a * HD + h];
}

__global__ void rbf_kernel(const float* __restrict__ pos, const int* __restrict__ erow,
                           const int* __restrict__ ecol, _Float16* __restrict__ rbf, int E) {
    int e = blockIdx.x * blockDim.x + threadIdx.x;
    if (e >= E) return;
    int a = erow[e], b = ecol[e];
    float dx = pos[a * 3 + 0] - pos[b * 3 + 0];
    float dy = pos[a * 3 + 1] - pos[b * 3 + 1];
    float dz = pos[a * 3 + 2] - pos[b * 3 + 2];
    float d = sqrtf(dx * dx + dy * dy + dz * dz);
    const float width = CUT / (float)RB;            // 0.1
    const float step  = CUT / (float)(RB - 1);      // linspace(0, 5, 50) step
    const float inv2w2 = 1.0f / (2.0f * width * width);
    #pragma unroll 8
    for (int j = 0; j < 64; ++j) {
        float v = 0.0f;
        if (j < RB) { float t = d - (float)j * step; v = __expf(-t * t * inv2w2); }
        rbf[(size_t)e * 64 + j] = (_Float16)v;
    }
}

// W1 [L, H+R, H] f32 -> W1T [L, H, KP] f16 (transposed, K-padded)
__global__ void conv_w1t(const float* __restrict__ W1, _Float16* __restrict__ W1T) {
    int i = blockIdx.x * blockDim.x + threadIdx.x;
    if (i >= NL * HD * KP) return;
    int l = i / (HD * KP); int r = i % (HD * KP);
    int col = r / KP; int k = r % KP;
    float v = (k < HD + RB) ? W1[((size_t)l * (HD + RB) + k) * HD + col] : 0.0f;
    W1T[i] = (_Float16)v;
}

// W2 [L, H, H] -> W2T [L, H, H] (transposed)
__global__ void conv_w2t(const float* __restrict__ W2, _Float16* __restrict__ W2T) {
    int i = blockIdx.x * blockDim.x + threadIdx.x;
    if (i >= NL * HD * HD) return;
    int l = i / (HD * HD); int r = i % (HD * HD);
    int col = r / HD; int k = r % HD;
    W2T[i] = (_Float16)W2[((size_t)l * HD + k) * HD + col];
}

// W_ih [L, H, 3H] -> WihT [L, 3H, H] (transposed)
__global__ void conv_wiht(const float* __restrict__ Wih, _Float16* __restrict__ WihT) {
    int i = blockIdx.x * blockDim.x + threadIdx.x;
    if (i >= NL * G3 * HD) return;
    int l = i / (G3 * HD); int r = i % (G3 * HD);
    int col = r / HD; int k = r % HD;
    WihT[i] = (_Float16)Wih[((size_t)l * HD + k) * G3 + col];
}

// ---------------- fragment loaders (wave32 WMMA layouts) ----------------
// A 16x32 f16: lanes 0-15/16-31 both hold M=lane&15; hi lanes take K+16..K+31.
__device__ __forceinline__ v16h load_a_lds(const _Float16* base, int row, int kstride,
                                           int kBase, int laneM, int laneHi) {
    AF a;
    const u32x4* pa = (const u32x4*)(base + (size_t)row * kstride + kBase + laneHi * 16);
    a.u[0] = pa[0];      // K = kBase + laneHi*16 .. +7
    a.u[1] = pa[1];      // K = kBase + laneHi*16 + 8 .. +15
    return a.v;
}
// B 32x16 f16 from transposed weights WT[outCol][K]: lane N = colBase+laneM,
// halves are 16 contiguous K values starting at kBase + laneHi*16.
__device__ __forceinline__ v16h load_b_wt(const _Float16* WT, int col, int kstride,
                                          int kBase, int laneHi) {
    AF b;
    const u32x4* pb = (const u32x4*)(WT + (size_t)col * kstride + kBase + laneHi * 16);
    b.u[0] = pb[0];
    b.u[1] = pb[1];
    return b.v;
}

__device__ __forceinline__ float silu(float v) { return v / (1.0f + __expf(-v)); }
__device__ __forceinline__ float sigmoidf_(float v) { return 1.0f / (1.0f + __expf(-v)); }

// ---------------- fused edge message kernel ----------------
// Per block: 64 edges x 128 outputs. GEMM1 [64,192]x[192,128] + SiLU (to LDS f16),
// then GEMM2 [64,128]x[128,128] + bias, scatter-add into aggr[col[e]].
__global__ __launch_bounds__(128) void edge_msg_kernel(
    const float* __restrict__ x,            // [N, H] f32
    const _Float16* __restrict__ rbf,       // [E, 64] f16 (padded)
    const int* __restrict__ erow,
    const int* __restrict__ ecol,
    const _Float16* __restrict__ W1T,       // [H, KP] f16 (this layer)
    const float* __restrict__ b1,           // [H]
    const _Float16* __restrict__ W2T,       // [H, H] f16
    const float* __restrict__ b2,           // [H]
    float* __restrict__ aggr,               // [N, H]
    int E)
{
    __shared__ __align__(16) _Float16 sInp[64 * KP];   // 24 KB
    __shared__ __align__(16) _Float16 sHid[64 * HD];   // 16 KB

    const int tid = threadIdx.x;
    const int e0  = blockIdx.x * 64;

    // cooperative gather + concat + f32->f16 convert into LDS
    for (int idx = tid; idx < 64 * KP; idx += 128) {
        int r = idx / KP, c = idx % KP;
        int e = e0 + r;
        _Float16 v = (_Float16)0.0f;
        if (e < E) {
            if (c < HD) v = (_Float16)x[(size_t)erow[e] * HD + c];
            else        v = rbf[(size_t)e * 64 + (c - HD)];
        }
        sInp[idx] = v;
    }
    __syncthreads();

    const int wave   = tid >> 5;
    const int lane   = tid & 31;
    const int laneM  = lane & 15;
    const int laneHi = lane >> 4;
    const int rowBase = wave * 16;

    // GEMM1 + bias + SiLU -> sHid (f16)
    #pragma unroll
    for (int n = 0; n < 8; ++n) {
        const int colBase = n * 16;
        v8f acc = {};
        #pragma unroll
        for (int k = 0; k < KP; k += 32) {
            v16h a = load_a_lds(sInp, rowBase + laneM, KP, k, laneM, laneHi);
            v16h b = load_b_wt(W1T, colBase + laneM, KP, k, laneHi);
            acc = __builtin_amdgcn_wmma_f32_16x16x32_f16(false, a, false, b,
                                                         (short)0, acc, false, false);
        }
        const float bias = b1[colBase + laneM];
        #pragma unroll
        for (int r = 0; r < 8; ++r) {
            float v = silu(acc[r] + bias);
            sHid[(size_t)(rowBase + laneHi * 8 + r) * HD + colBase + laneM] = (_Float16)v;
        }
    }
    __syncthreads();

    // GEMM2 + bias -> atomic scatter into aggr
    #pragma unroll
    for (int n = 0; n < 8; ++n) {
        const int colBase = n * 16;
        v8f acc = {};
        #pragma unroll
        for (int k = 0; k < HD; k += 32) {
            v16h a = load_a_lds(sHid, rowBase + laneM, HD, k, laneM, laneHi);
            v16h b = load_b_wt(W2T, colBase + laneM, HD, k, laneHi);
            acc = __builtin_amdgcn_wmma_f32_16x16x32_f16(false, a, false, b,
                                                         (short)0, acc, false, false);
        }
        const float bias = b2[colBase + laneM];
        #pragma unroll
        for (int r = 0; r < 8; ++r) {
            int e = e0 + rowBase + laneHi * 8 + r;
            if (e < E) {
                atomicAdd(&aggr[(size_t)ecol[e] * HD + colBase + laneM], acc[r] + bias);
            }
        }
    }
}

// ---------------- GRU input GEMM: gi = aggr @ W_ih + b_ih ----------------
__global__ __launch_bounds__(128) void gru_gemm_kernel(
    const float* __restrict__ aggr,        // [N, H]
    const _Float16* __restrict__ WihT,     // [3H, H] (this layer)
    const float* __restrict__ bih,         // [3H]
    float* __restrict__ gi,                // [N, 3H]
    int N)
{
    __shared__ __align__(16) _Float16 sA[64 * HD];     // 16 KB

    const int tid = threadIdx.x;
    const int n0  = blockIdx.x * 64;

    for (int idx = tid; idx < 64 * HD; idx += 128) {
        int r = idx / HD, c = idx % HD;
        int node = n0 + r;
        sA[idx] = (node < N) ? (_Float16)aggr[(size_t)node * HD + c] : (_Float16)0.0f;
    }
    __syncthreads();

    const int wave   = tid >> 5;
    const int lane   = tid & 31;
    const int laneM  = lane & 15;
    const int laneHi = lane >> 4;
    const int rowBase = wave * 16;

    for (int n = 0; n < G3 / 16; ++n) {               // 24 output col-tiles
        const int colBase = n * 16;
        v8f acc = {};
        #pragma unroll
        for (int k = 0; k < HD; k += 32) {
            v16h a = load_a_lds(sA, rowBase + laneM, HD, k, laneM, laneHi);
            v16h b = load_b_wt(WihT, colBase + laneM, HD, k, laneHi);
            acc = __builtin_amdgcn_wmma_f32_16x16x32_f16(false, a, false, b,
                                                         (short)0, acc, false, false);
        }
        const float bias = bih[colBase + laneM];
        #pragma unroll
        for (int r = 0; r < 8; ++r) {
            int node = n0 + rowBase + laneHi * 8 + r;
            if (node < N) gi[(size_t)node * G3 + colBase + laneM] = acc[r] + bias;
        }
    }
}

// ---------------- GRU elementwise update (h0 = 0) ----------------
__global__ void gru_update_kernel(const float* __restrict__ gi,
                                  const float* __restrict__ bhh,   // [3H] this layer
                                  float* __restrict__ x, int N)
{
    int i = blockIdx.x * blockDim.x + threadIdx.x;
    if (i >= N * HD) return;
    int n = i / HD, h = i % HD;
    float gr = gi[(size_t)n * G3 + h];
    float gz = gi[(size_t)n * G3 + HD + h];
    float gn = gi[(size_t)n * G3 + 2 * HD + h];
    float r  = sigmoidf_(gr + bhh[h]);
    float z  = sigmoidf_(gz + bhh[HD + h]);
    float nn = tanhf(gn + r * bhh[2 * HD + h]);
    x[i] += (1.0f - z) * nn;
}

// ---------------- LayerNorm + pooled atomics ----------------
__global__ __launch_bounds__(128) void ln_pool_kernel(
    const float* __restrict__ x, const float* __restrict__ g, const float* __restrict__ bt,
    const int* __restrict__ batch, float* __restrict__ out,
    float* __restrict__ sums, float* __restrict__ counts, int N)
{
    __shared__ float red[HD];
    int n = blockIdx.x;
    int h = threadIdx.x;
    if (n >= N) return;
    float v = x[(size_t)n * HD + h];
    red[h] = v;
    __syncthreads();
    for (int s = HD / 2; s > 0; s >>= 1) {
        if (h < s) red[h] += red[h + s];
        __syncthreads();
    }
    float mu = red[0] * (1.0f / HD);
    __syncthreads();
    float dv = v - mu;
    red[h] = dv * dv;
    __syncthreads();
    for (int s = HD / 2; s > 0; s >>= 1) {
        if (h < s) red[h] += red[h + s];
        __syncthreads();
    }
    float var = red[0] * (1.0f / HD);
    float y = dv * rsqrtf(var + LN_EPS) * g[h] + bt[h];
    out[(size_t)n * HD + h] = y;
    int m = batch[n];
    atomicAdd(&sums[(size_t)m * HD + h], y);
    if (h == 0) atomicAdd(&counts[m], 1.0f);
}

__global__ void graph_mean_kernel(const float* __restrict__ sums,
                                  const float* __restrict__ counts,
                                  float* __restrict__ out)
{
    int i = blockIdx.x * blockDim.x + threadIdx.x;
    if (i >= NM * HD) return;
    int m = i / HD;
    float c = counts[m];
    out[i] = sums[i] / (c < 1.0f ? 1.0f : c);
}

// ---------------- host launch ----------------
extern "C" void kernel_launch(void* const* d_in, const int* in_sizes, int n_in,
                              void* d_out, int out_size, void* d_ws, size_t ws_size,
                              hipStream_t stream) {
    (void)n_in; (void)out_size; (void)ws_size;
    const int*   an    = (const int*)  d_in[0];
    const float* pos   = (const float*)d_in[1];
    const int*   batch = (const int*)  d_in[2];
    const int*   ei    = (const int*)  d_in[3];
    const float* emb   = (const float*)d_in[4];
    const float* W1    = (const float*)d_in[5];
    const float* b1    = (const float*)d_in[6];
    const float* W2    = (const float*)d_in[7];
    const float* b2    = (const float*)d_in[8];
    const float* Wih   = (const float*)d_in[9];
    const float* bih   = (const float*)d_in[10];
    const float* bhh   = (const float*)d_in[11];
    const float* lng   = (const float*)d_in[12];
    const float* lnb   = (const float*)d_in[13];

    const int N = in_sizes[0];
    const int E = in_sizes[3] / 2;
    const int* erow = ei;
    const int* ecol = ei + E;

    char* p = (char*)d_ws;
    auto take = [&](size_t bytes) -> char* {
        char* q = p; p += (bytes + 255) & ~(size_t)255; return q;
    };
    float*    x     = (float*)   take((size_t)N * HD * 4);
    float*    aggr  = (float*)   take((size_t)N * HD * 4);
    float*    gi    = (float*)   take((size_t)N * G3 * 4);
    _Float16* rbf   = (_Float16*)take((size_t)E * 64 * 2);
    _Float16* W1T   = (_Float16*)take((size_t)NL * HD * KP * 2);
    _Float16* W2T   = (_Float16*)take((size_t)NL * HD * HD * 2);
    _Float16* WihT  = (_Float16*)take((size_t)NL * G3 * HD * 2);
    float*    sums  = (float*)   take((size_t)NM * HD * 4);
    float*    cnts  = (float*)   take((size_t)NM * 4);

    float* out_x = (float*)d_out;
    float* out_g = out_x + (size_t)N * HD;

    // one-time-per-call prep (deterministic)
    conv_w1t <<<(NL * HD * KP + 255) / 256, 256, 0, stream>>>(W1, W1T);
    conv_w2t <<<(NL * HD * HD + 255) / 256, 256, 0, stream>>>(W2, W2T);
    conv_wiht<<<(NL * G3 * HD + 255) / 256, 256, 0, stream>>>(Wih, WihT);
    embed_kernel<<<(N * HD + 255) / 256, 256, 0, stream>>>(an, emb, x, N);
    if (E > 0)
        rbf_kernel<<<(E + 255) / 256, 256, 0, stream>>>(pos, erow, ecol, rbf, E);
    zero_f32<<<(NM * HD + 255) / 256, 256, 0, stream>>>(sums, NM * HD);
    zero_f32<<<1, 64, 0, stream>>>(cnts, NM);

    for (int l = 0; l < NL; ++l) {
        zero_f32<<<(N * HD + 255) / 256, 256, 0, stream>>>(aggr, N * HD);
        if (E > 0)
            edge_msg_kernel<<<(E + 63) / 64, 128, 0, stream>>>(
                x, rbf, erow, ecol,
                W1T + (size_t)l * HD * KP, b1 + (size_t)l * HD,
                W2T + (size_t)l * HD * HD, b2 + (size_t)l * HD,
                aggr, E);
        gru_gemm_kernel<<<(N + 63) / 64, 128, 0, stream>>>(
            aggr, WihT + (size_t)l * G3 * HD, bih + (size_t)l * G3, gi, N);
        gru_update_kernel<<<(N * HD + 255) / 256, 256, 0, stream>>>(
            gi, bhh + (size_t)l * G3, x, N);
    }

    ln_pool_kernel<<<N, HD, 0, stream>>>(x, lng, lnb, batch, out_x, sums, cnts, N);
    graph_mean_kernel<<<(NM * HD + 255) / 256, 256, 0, stream>>>(sums, cnts, out_g);
}
